// TA2N_26328149524501
// MI455X (gfx1250) — matrix operand order repaired and checked by reading, not compiled, not run on hardware
//
#include <hip/hip_runtime.h>
#include <hip/hip_bf16.h>

typedef __attribute__((ext_vector_type(16))) _Float16 v16h;
typedef __attribute__((ext_vector_type(8)))  _Float16 v8h;
typedef __attribute__((ext_vector_type(8)))  float    v8f;

#define CDIV(a,b) (((a)+(b)-1)/(b))

__constant__ float PXC[9] = {0.f, 0.f, .2f, 0.f, -.2f, .2f, -.2f, .2f, -.2f};
__constant__ float PYC[9] = {0.f, .2f, 0.f, -.2f, 0.f, .2f, -.2f, -.2f, .2f};

// ---------------------------------------------------------------- WMMA helpers
__device__ inline v8f wmma_f16(v16h a, v16h b, v8f c) {
  return __builtin_amdgcn_wmma_f32_16x16x32_f16(false, a, false, b, (short)0, c,
                                                false, false);
}

// A fragment (16x32 MxK, f16) from LDS row-major [16][ld] tile.
// ISA 7.12.2: lane m=L&15, half=L>>4; VGPR v<4 -> K=8h+2v ; v>=4 -> 16+8h+2(v-4)
// => f[0..7] = K 8h..8h+7, f[8..15] = K 16+8h..16+8h+7 : two aligned 16B chunks.
__device__ inline v16h frag_a16(const _Float16* base, int ld) {
  int lane = threadIdx.x & 31;
  int r = lane & 15, half = lane >> 4;
  v8h lo = *(const v8h*)(base + r * ld + half * 8);
  v8h hi = *(const v8h*)(base + r * ld + 16 + half * 8);
  return __builtin_shufflevector(lo, hi, 0, 1, 2, 3, 4, 5, 6, 7,
                                 8, 9, 10, 11, 12, 13, 14, 15);
}

// B fragment (32x16 KxN, f16) from LDS N-major transposed [16][ld] tile.
// lane n=L&15, half=L>>4; f[0..7] = K 8h..8h+7, f[8..15] = K 16+8h..16+8h+7.
__device__ inline v16h frag_b16(const _Float16* base, int ld) {
  int lane = threadIdx.x & 31;
  int r = lane & 15, half = lane >> 4;
  v8h lo = *(const v8h*)(base + r * ld + half * 8);
  v8h hi = *(const v8h*)(base + r * ld + 16 + half * 8);
  return __builtin_shufflevector(lo, hi, 0, 1, 2, 3, 4, 5, 6, 7,
                                 8, 9, 10, 11, 12, 13, 14, 15);
}

// ------------------------------------------------------- implicit-GEMM conv3d
// MODE 0: locnet c1, samples 0..4 from src0(support), 5..14 from src1(query)
// MODE 1: plain contiguous tensor src0 (NS,Cin,T,H,H)
// MODE 2: acm c1 virtual pairs_in: c<256 -> s_pr[n], c>=256 -> q_al[m]+adj[n,m,c,t]
// Tile: 64(M) x 128(N), Ktile 32, 8 waves, each wave 32x32 via 4 WMMA.
template <int MODE>
__global__ __launch_bounds__(256) void conv3d_wmma(
    const float* __restrict__ Wt, const float* __restrict__ cb,
    const float* __restrict__ gg, const float* __restrict__ bb,
    const float* __restrict__ src0, const float* __restrict__ src1,
    const float* __restrict__ adj, float* __restrict__ out,
    int Cout, int Cin, int T, int H, int NS) {
  const int Ktot = Cin * 27;  // always a multiple of 32 here
  const int HH = H * H;
  const int L = T * HH;
  const int Ntot = NS * L;
  __shared__ __align__(16) _Float16 As[64 * 40];
  __shared__ __align__(16) _Float16 Bt[128 * 40];
  const int tid = threadIdx.x;
  const int wave = tid >> 5, lane = tid & 31;
  const int wm = wave >> 2, wn = wave & 3;
  const int bM = blockIdx.x, bN = blockIdx.y;

  // ---- per-thread B-column decode, hoisted out of the K loop (2 rows/thread)
  int nrow[2], pp[2], tt[2], yy0[2], xx0[2];
  const float* basep[2];   // MODE-specific primary base
  const float* baseq[2];   // MODE 2 query base
  const float* basea[2];   // MODE 2 adj base
#pragma unroll
  for (int rep = 0; rep < 2; ++rep) {
    int nr = rep * 64 + (tid >> 2);
    nrow[rep] = nr;
    int gn = bN * 128 + nr;
    int gnc = gn < Ntot ? gn : 0;
    int p = gnc / L; int r = gnc - p * L;
    int t = r / HH; r -= t * HH;
    pp[rep] = p; tt[rep] = t; yy0[rep] = r / H; xx0[rep] = r - (r / H) * H;
    if (MODE == 0) {
      basep[rep] = (p < 5) ? (src0 + (size_t)p * Cin * L)
                           : (src1 + (size_t)(p - 5) * Cin * L);
      baseq[rep] = basep[rep]; basea[rep] = adj;
    } else if (MODE == 1) {
      basep[rep] = src0 + (size_t)p * Cin * L;
      baseq[rep] = basep[rep]; basea[rep] = adj;
    } else {
      int sn = p / 10, qm = p - sn * 10;
      basep[rep] = src0 + (size_t)sn * 256 * L;
      baseq[rep] = src1 + (size_t)qm * 256 * L;
      basea[rep] = adj + ((size_t)sn * 10 + qm) * 256 * T;
    }
  }
  const int arow = tid >> 2;            // A staging row (64 rows, 4 thr/row)
  const int akc  = (tid & 3) * 8;      // 8 consecutive K
  const int garow = bM * 64 + arow;

  v8f acc[2][2];
#pragma unroll
  for (int i = 0; i < 2; ++i)
#pragma unroll
    for (int j = 0; j < 2; ++j) acc[i][j] = v8f{0.f,0.f,0.f,0.f,0.f,0.f,0.f,0.f};

  for (int k0 = 0; k0 < Ktot; k0 += 32) {
    {  // ---- stage A (weights OIDHW == row-major MxK), vector f32 loads
      v8h av;
      if (garow < Cout) {
        const float4* wp = (const float4*)(Wt + (size_t)garow * Ktot + k0 + akc);
        float4 w0 = wp[0], w1 = wp[1];
        av[0] = (_Float16)w0.x; av[1] = (_Float16)w0.y;
        av[2] = (_Float16)w0.z; av[3] = (_Float16)w0.w;
        av[4] = (_Float16)w1.x; av[5] = (_Float16)w1.y;
        av[6] = (_Float16)w1.z; av[7] = (_Float16)w1.w;
        if (k0 + 32 < Ktot)
          __builtin_prefetch(Wt + (size_t)garow * Ktot + k0 + 32 + akc, 0, 1);
      } else {
#pragma unroll
        for (int i = 0; i < 8; ++i) av[i] = (_Float16)0.f;
      }
      *(v8h*)(As + arow * 40 + akc) = av;
    }
    {  // ---- stage B (im2col gather), branchless padding, N-major store
#pragma unroll
      for (int rep = 0; rep < 2; ++rep) {
        int gn = bN * 128 + nrow[rep];
        v8h bv;
        if (gn < Ntot) {
          int kc = (tid & 3) * 8;
#pragma unroll
          for (int i = 0; i < 8; ++i) {
            int gk = k0 + kc + i;
            int c = gk / 27; int kk = gk - c * 27;
            int kd = kk / 9; kk -= kd * 9;
            int kh = kk / 3; int kw2 = kk - kh * 3;
            int td = tt[rep] + kd - 1, ya = yy0[rep] + kh - 1, xa = xx0[rep] + kw2 - 1;
            float flag = (((unsigned)td < (unsigned)T) &
                          ((unsigned)ya < (unsigned)H) &
                          ((unsigned)xa < (unsigned)H)) ? 1.f : 0.f;
            int tdc = td < 0 ? 0 : (td >= T ? T - 1 : td);
            int yac = ya < 0 ? 0 : (ya >= H ? H - 1 : ya);
            int xac = xa < 0 ? 0 : (xa >= H ? H - 1 : xa);
            size_t sp = (size_t)tdc * HH + yac * H + xac;
            float v;
            if (MODE == 2) {
              bool qs = c >= 256;
              int cq = qs ? (c - 256) : c;
              const float* bp = qs ? baseq[rep] : basep[rep];
              float ev = basea[rep][(size_t)cq * T + tdc];  // safe unconditional
              v = flag * (bp[(size_t)cq * L + sp] + (qs ? ev : 0.f));
            } else {
              v = flag * basep[rep][(size_t)c * L + sp];
            }
            bv[i] = (_Float16)v;
          }
        } else {
#pragma unroll
          for (int i = 0; i < 8; ++i) bv[i] = (_Float16)0.f;
        }
        *(v8h*)(Bt + nrow[rep] * 40 + (tid & 3) * 8) = bv;
      }
    }
    __syncthreads();
    v16h a0 = frag_a16(As + (wm * 32) * 40, 40);
    v16h a1 = frag_a16(As + (wm * 32 + 16) * 40, 40);
    v16h b0 = frag_b16(Bt + (wn * 32) * 40, 40);
    v16h b1 = frag_b16(Bt + (wn * 32 + 16) * 40, 40);
    acc[0][0] = wmma_f16(a0, b0, acc[0][0]);
    acc[0][1] = wmma_f16(a0, b1, acc[0][1]);
    acc[1][0] = wmma_f16(a1, b0, acc[1][0]);
    acc[1][1] = wmma_f16(a1, b1, acc[1][1]);
    __syncthreads();
  }
  // epilogue: fused eval BN: y = conv*s + (conv_bias*s + beta), s = g/sqrt(1+1e-5)
  const int n = lane & 15, mb = (lane >> 4) * 8;
#pragma unroll
  for (int am = 0; am < 2; ++am) {
#pragma unroll
    for (int i = 0; i < 8; ++i) {
      int m = bM * 64 + wm * 32 + am * 16 + mb + i;
      if (m >= Cout) continue;
      float sc = gg[m] * 0.9999950000374997f;
      float bi = cb[m] * sc + bb[m];
#pragma unroll
      for (int bn = 0; bn < 2; ++bn) {
        int gn = bN * 128 + wn * 32 + bn * 16 + n;
        if (gn >= Ntot) continue;
        float v = acc[am][bn][i] * sc + bi;
        int p = gn / L; int r = gn - p * L;
        out[((size_t)p * Cout + m) * L + r] = v;
      }
    }
  }
}

// ---------------------------------------------- batched channel GEMM (1x1 conv)
// out[b,M,L] = W[M,K] @ X[b,K,L] (+bias)(relu). Same 64x128 WMMA tile.
__global__ __launch_bounds__(256) void gemm_wmma(
    const float* __restrict__ Wt, const float* __restrict__ X,
    const float* __restrict__ bias, float* __restrict__ out,
    int M, int K, int B, int L, int relu) {
  const int Ntot = B * L;
  __shared__ __align__(16) _Float16 As[64 * 40];
  __shared__ __align__(16) _Float16 Bt[128 * 40];
  const int tid = threadIdx.x;
  const int wave = tid >> 5, lane = tid & 31;
  const int wm = wave >> 2, wn = wave & 3;
  const int bM = blockIdx.x, bN = blockIdx.y;

  const float* bbase[2];
  int nrow[2];
#pragma unroll
  for (int rep = 0; rep < 2; ++rep) {
    int nr = rep * 64 + (tid >> 2);
    nrow[rep] = nr;
    int gn = bN * 128 + nr;
    int gnc = gn < Ntot ? gn : 0;
    int b = gnc / L, l = gnc - (gnc / L) * L;
    bbase[rep] = X + (size_t)b * K * L + l;
  }
  const int arow = tid >> 2, akc = (tid & 3) * 8;
  const int garow = bM * 64 + arow;

  v8f acc[2][2];
#pragma unroll
  for (int i = 0; i < 2; ++i)
#pragma unroll
    for (int j = 0; j < 2; ++j) acc[i][j] = v8f{0.f,0.f,0.f,0.f,0.f,0.f,0.f,0.f};

  for (int k0 = 0; k0 < K; k0 += 32) {
    {
      v8h av;
      if (garow < M) {
        const float4* wp = (const float4*)(Wt + (size_t)garow * K + k0 + akc);
        float4 w0 = wp[0], w1 = wp[1];
        av[0] = (_Float16)w0.x; av[1] = (_Float16)w0.y;
        av[2] = (_Float16)w0.z; av[3] = (_Float16)w0.w;
        av[4] = (_Float16)w1.x; av[5] = (_Float16)w1.y;
        av[6] = (_Float16)w1.z; av[7] = (_Float16)w1.w;
      } else {
#pragma unroll
        for (int i = 0; i < 8; ++i) av[i] = (_Float16)0.f;
      }
      *(v8h*)(As + arow * 40 + akc) = av;
    }
#pragma unroll
    for (int rep = 0; rep < 2; ++rep) {
      int gn = bN * 128 + nrow[rep];
      v8h bv;
      if (gn < Ntot) {
#pragma unroll
        for (int i = 0; i < 8; ++i)
          bv[i] = (_Float16)bbase[rep][(size_t)(k0 + akc + i) * L];
      } else {
#pragma unroll
        for (int i = 0; i < 8; ++i) bv[i] = (_Float16)0.f;
      }
      *(v8h*)(Bt + nrow[rep] * 40 + akc) = bv;
    }
    __syncthreads();
    v16h a0 = frag_a16(As + (wm * 32) * 40, 40);
    v16h a1 = frag_a16(As + (wm * 32 + 16) * 40, 40);
    v16h b0 = frag_b16(Bt + (wn * 32) * 40, 40);
    v16h b1 = frag_b16(Bt + (wn * 32 + 16) * 40, 40);
    acc[0][0] = wmma_f16(a0, b0, acc[0][0]);
    acc[0][1] = wmma_f16(a0, b1, acc[0][1]);
    acc[1][0] = wmma_f16(a1, b0, acc[1][0]);
    acc[1][1] = wmma_f16(a1, b1, acc[1][1]);
    __syncthreads();
  }
  const int n = lane & 15, mb = (lane >> 4) * 8;
#pragma unroll
  for (int am = 0; am < 2; ++am) {
#pragma unroll
    for (int i = 0; i < 8; ++i) {
      int m = bM * 64 + wm * 32 + am * 16 + mb + i;
      if (m >= M) continue;
      float bi = bias ? bias[m] : 0.f;
#pragma unroll
      for (int bn = 0; bn < 2; ++bn) {
        int gn = bN * 128 + wn * 32 + bn * 16 + n;
        if (gn >= Ntot) continue;
        float v = acc[am][bn][i] + bi;
        if (relu) v = fmaxf(v, 0.f);
        int b = gn / L, l = gn - (gn / L) * L;
        out[((size_t)b * M + m) * L + l] = v;
      }
    }
  }
}

// ------------------------------------------------------------- small kernels
__global__ void locnet_pool1_kernel(const float* __restrict__ in,
                                    float* __restrict__ out, int total) {
  int idx = blockIdx.x * blockDim.x + threadIdx.x;
  if (idx >= total) return;  // (15,64,4,3,3)
  int x = idx % 3; int r = idx / 3;
  int y = r % 3;   r /= 3;
  int td = r % 4;  int bc = r / 4;  // 15*64
  const float* p = in + (size_t)bc * 392;
  float mx = -1e30f;
  for (int dd = 0; dd < 2; ++dd)
    for (int hh = 0; hh < 2; ++hh)
      for (int ww = 0; ww < 2; ++ww)
        mx = fmaxf(mx, p[(2 * td + dd) * 49 + (2 * y + hh) * 7 + (2 * x + ww)]);
  out[idx] = fmaxf(mx, 0.f);
}

__global__ void locnet_tail_kernel(const float* __restrict__ h2,
                                   const float* __restrict__ l1w,
                                   const float* __restrict__ l1b,
                                   const float* __restrict__ l2w,
                                   const float* __restrict__ l2b,
                                   float* __restrict__ theta) {
  __shared__ float f[128];
  __shared__ float g1[32];
  int s = blockIdx.x, tid = threadIdx.x;
  if (tid < 128) {  // pool(2,2,2)+relu+globalmax == max over t0..3,y0..1,x0..1, relu
    const float* p = h2 + ((size_t)s * 128 + tid) * 36;
    float mx = -1e30f;
    for (int t = 0; t < 4; ++t)
      for (int y = 0; y < 2; ++y)
        for (int x = 0; x < 2; ++x) mx = fmaxf(mx, p[t * 9 + y * 3 + x]);
    f[tid] = fmaxf(mx, 0.f);
  }
  __syncthreads();
  if (tid < 32) {
    float a = l1b[tid];
    for (int j = 0; j < 128; ++j) a += l1w[tid * 128 + j] * f[j];
    g1[tid] = fmaxf(a, 0.f);
  }
  __syncthreads();
  if (tid < 2) {
    float a = l2b[tid];
    for (int j = 0; j < 32; ++j) a += l2w[tid * 32 + j] * g1[j];
    theta[s * 2 + tid] = tanhf(a);
  }
}

__global__ void warp_kernel(const float* __restrict__ support,
                            const float* __restrict__ query,
                            const float* __restrict__ theta,
                            float* __restrict__ s_al, float* __restrict__ q_al,
                            int total) {
  int idx = blockIdx.x * blockDim.x + threadIdx.x;
  if (idx >= total) return;  // idx = ((s*256+c)*8+t)*49+hw
  int hw = idx % 49; int r = idx / 49;
  int t = r & 7; r >>= 3;
  int c = r & 255; int s = r >> 8;
  float th0 = theta[s * 2], th1 = theta[s * 2 + 1];
  float base = -1.f + t * (2.f / 7.f);
  float g = th0 * base + th1;
  float xf = (g + 1.f) * 3.5f;  // align_corners unnormalize, T=8
  float x0 = floorf(xf);
  float w1 = xf - x0;
  int i0 = (int)x0, i1 = i0 + 1;
  float v0 = (i0 >= 0 && i0 <= 7) ? 1.f : 0.f;
  float v1 = (i1 >= 0 && i1 <= 7) ? 1.f : 0.f;
  int c0 = i0 < 0 ? 0 : (i0 > 7 ? 7 : i0);
  int c1 = i1 < 0 ? 0 : (i1 > 7 ? 7 : i1);
  const float* src; float* dst;
  if (s < 5) { src = support + (size_t)s * 100352; dst = s_al + (size_t)s * 100352; }
  else       { src = query + (size_t)(s - 5) * 100352; dst = q_al + (size_t)(s - 5) * 100352; }
  float val = src[(c * 8 + c0) * 49 + hw] * ((1.f - w1) * v0) +
              src[(c * 8 + c1) * 49 + hw] * (w1 * v1);
  dst[(c * 8 + t) * 49 + hw] = val;
}

__global__ void mean_hw_kernel(const float* __restrict__ in,
                               float* __restrict__ out, int total, int HH) {
  int idx = blockIdx.x * blockDim.x + threadIdx.x;
  if (idx >= total) return;
  const float* p = in + (size_t)idx * HH;
  float a = 0.f;
  for (int i = 0; i < HH; ++i) a += p[i];
  out[idx] = a / (float)HH;
}

// softmax over x (insupport): attn[n,x,y]
__global__ void attn1_kernel(const float* __restrict__ keys,
                             const float* __restrict__ qrs,
                             float* __restrict__ attn) {
  __shared__ float A[64];
  int nidx = blockIdx.x, tid = threadIdx.x;
  if (tid < 64) {
    int x = tid >> 3, y = tid & 7;
    const float* kp = keys + (size_t)nidx * 2048;
    const float* qp = qrs + (size_t)nidx * 2048;
    float a = 0.f;
    for (int c = 0; c < 256; ++c) a += kp[c * 8 + x] * qp[c * 8 + y];
    A[x * 8 + y] = a * 0.0625f;
  }
  __syncthreads();
  if (tid < 8) {
    int y = tid;
    float mx = -1e30f;
    for (int x = 0; x < 8; ++x) mx = fmaxf(mx, A[x * 8 + y]);
    float e[8], s = 0.f;
    for (int x = 0; x < 8; ++x) { e[x] = __expf(A[x * 8 + y] - mx); s += e[x]; }
    for (int x = 0; x < 8; ++x) attn[(size_t)nidx * 64 + x * 8 + y] = e[x] / s;
  }
}

// softmax over y (acm): attn[n,m,x,y], block = n*10+m
__global__ void attn2_kernel(const float* __restrict__ keys,
                             const float* __restrict__ qrs,
                             float* __restrict__ attn) {
  __shared__ float A[64];
  int b = blockIdx.x, tid = threadIdx.x;
  int n = b / 10, m = b - n * 10;
  if (tid < 64) {
    int x = tid >> 3, y = tid & 7;
    const float* kp = keys + (size_t)n * 2048;
    const float* qp = qrs + (size_t)m * 2048;
    float a = 0.f;
    for (int c = 0; c < 256; ++c) a += kp[c * 8 + x] * qp[c * 8 + y];
    A[x * 8 + y] = a * 0.0625f;
  }
  __syncthreads();
  if (tid < 8) {
    int x = tid;
    float mx = -1e30f;
    for (int y = 0; y < 8; ++y) mx = fmaxf(mx, A[x * 8 + y]);
    float e[8], s = 0.f;
    for (int y = 0; y < 8; ++y) { e[y] = __expf(A[x * 8 + y] - mx); s += e[y]; }
    for (int y = 0; y < 8; ++y) attn[(size_t)b * 64 + x * 8 + y] = e[y] / s;
  }
}

// proto[n,c,y,hw] = sum_x attn1[n,x,y] * vals[n,c,x,hw]
__global__ void proto_kernel(const float* __restrict__ attn,
                             const float* __restrict__ vals,
                             float* __restrict__ proto, int total) {
  int idx = blockIdx.x * blockDim.x + threadIdx.x;
  if (idx >= total) return;
  int hw = idx % 49; int r = idx / 49;
  int y = r & 7; r >>= 3;
  int c = r & 255; int n = r >> 8;
  float a = 0.f;
  for (int x = 0; x < 8; ++x)
    a += attn[(size_t)n * 64 + x * 8 + y] *
         vals[((size_t)n * 256 + c) * 392 + x * 49 + hw];
  proto[idx] = a;
}

// s_pr = proto + vals_s broadcast over hw
__global__ void spr_kernel(const float* __restrict__ proto,
                           const float* __restrict__ valss,
                           float* __restrict__ s_pr, int total) {
  int idx = blockIdx.x * blockDim.x + threadIdx.x;
  if (idx >= total) return;
  int r = idx / 49;
  int t = r & 7; r >>= 3;
  int c = r & 255; int n = r >> 8;
  s_pr[idx] = proto[idx] + valss[((size_t)n * 256 + c) * 8 + t];
}

// adj[n,m,c,x] = sum_y attn2[n,m,x,y] * valsq[m,c,y]
__global__ void adj_kernel(const float* __restrict__ attn,
                           const float* __restrict__ valsq,
                           float* __restrict__ adj, int total) {
  int idx = blockIdx.x * blockDim.x + threadIdx.x;
  if (idx >= total) return;
  int x = idx & 7; int r = idx >> 3;
  int c = r & 255; r >>= 8;
  int m = r % 10;  int n = r / 10;
  float a = 0.f;
  for (int y = 0; y < 8; ++y)
    a += attn[((size_t)n * 10 + m) * 64 + x * 8 + y] *
         valsq[((size_t)m * 256 + c) * 8 + y];
  adj[idx] = a;
}

__global__ void mv_pool1_kernel(const float* __restrict__ in,
                                float* __restrict__ out, int total) {
  int idx = blockIdx.x * blockDim.x + threadIdx.x;
  if (idx >= total) return;  // (50,128,8,3,3), window (1,2,2)
  int x = idx % 3; int r = idx / 3;
  int y = r % 3;   int bct = r / 3;  // 50*128*8
  const float* p = in + (size_t)bct * 49 + (2 * y) * 7 + 2 * x;
  float mx = fmaxf(fmaxf(p[0], p[1]), fmaxf(p[7], p[8]));
  out[idx] = fmaxf(mx, 0.f);
}

__global__ void mv_pool2_kernel(const float* __restrict__ in,
                                float* __restrict__ out, int total) {
  int idx = blockIdx.x * blockDim.x + threadIdx.x;
  if (idx >= total) return;  // (50,128,8) from (…,3,3)
  const float* p = in + (size_t)idx * 9;
  float mx = fmaxf(fmaxf(p[0], p[1]), fmaxf(p[3], p[4]));
  out[idx] = fmaxf(mx, 0.f);
}

__global__ void p2_kernel(const float* __restrict__ w, const float* __restrict__ bias,
                          const float* __restrict__ h, float* __restrict__ raw,
                          int total) {
  int idx = blockIdx.x * blockDim.x + threadIdx.x;
  if (idx >= total) return;  // (50,8,2)
  int o = idx & 1; int r = idx >> 1;
  int t = r & 7;   int b = r >> 3;
  float a = bias[o];
  for (int c = 0; c < 64; ++c) a += w[o * 64 + c] * h[((size_t)b * 64 + c) * 8 + t];
  raw[idx] = tanhf(a) * 0.75f;
}

// soft-mask pooling, one block per (n,m,t); writes both halves of pairs output
__global__ __launch_bounds__(256) void maskpool_kernel(
    const float* __restrict__ s_pr, const float* __restrict__ q_al,
    const float* __restrict__ adj, const float* __restrict__ raw,
    float* __restrict__ pairs) {
  __shared__ float ms[9 * 49], mq[9 * 49], avs[49], avq[49];
  int b = blockIdx.x;
  int t = b & 7; int p = b >> 3;
  int n = p / 10, m = p - n * 10;
  int tid = threadIdx.x;
  float ox0 = raw[(p * 8 + t) * 2 + 0];
  float oy0 = raw[(p * 8 + t) * 2 + 1];
  if (tid < 9 || (tid >= 16 && tid < 25)) {
    int j = (tid < 9) ? tid : tid - 16;
    float sgn = (tid < 9) ? 1.f : -1.f;
    float ox = sgn * (ox0 + PXC[j]);
    float oy = sgn * (oy0 + PYC[j]);
    float mx[7], my[7];
    for (int l = 0; l < 7; ++l) {
      float gl = l * (1.f / 3.f) - 1.f;
      float gx = gl + ox, gy = gl + oy;
      mx[l] = (gx > -1.f && gx < 1.f) ? 1.f : fmaxf(0.f, 1.f - (fabsf(gx) - 1.f) * 3.f);
      my[l] = (gy > -1.f && gy < 1.f) ? 1.f : fmaxf(0.f, 1.f - (fabsf(gy) - 1.f) * 3.f);
    }
    float sum = 0.f;
    for (int y = 0; y < 7; ++y)
      for (int x = 0; x < 7; ++x) sum += my[y] * mx[x];
    float inv = 1.f / sum;
    float* dstm = (tid < 9) ? ms : mq;
    for (int y = 0; y < 7; ++y)
      for (int x = 0; x < 7; ++x) dstm[j * 49 + y * 7 + x] = my[y] * mx[x] * inv;
  }
  __syncthreads();
  if (tid < 49) {
    float a = 0.f;
    for (int j = 0; j < 9; ++j) a += ms[j * 49 + tid];
    avs[tid] = a * (1.f / 9.f);
  } else if (tid >= 64 && tid < 113) {
    int i = tid - 64;
    float a = 0.f;
    for (int j = 0; j < 9; ++j) a += mq[j * 49 + i];
    avq[i] = a * (1.f / 9.f);
  }
  __syncthreads();
  int c = tid;  // 256 channels
  const float* sp = s_pr + ((size_t)n * 256 + c) * 392 + t * 49;
  const float* qp = q_al + ((size_t)m * 256 + c) * 392 + t * 49;
  float as = 0.f, aq = 0.f;
  for (int i = 0; i < 49; ++i) { as += avs[i] * sp[i]; aq += avq[i] * qp[i]; }
  aq += adj[(((size_t)n * 10 + m) * 256 + c) * 8 + t];  // mask sums to 1
  size_t ob = ((size_t)p * 512 + c) * 8 + t;
  pairs[ob] = as;
  pairs[ob + 2048] = aq;  // channel + 256
}

// -------------------------------------------------------------------- driver
extern "C" void kernel_launch(void* const* d_in, const int* in_sizes, int n_in,
                              void* d_out, int out_size, void* d_ws, size_t ws_size,
                              hipStream_t stream) {
  const float* support = (const float*)d_in[0];
  const float* query   = (const float*)d_in[1];
  const float* t_c1w = (const float*)d_in[2];  const float* t_c1b = (const float*)d_in[3];
  const float* t_g1  = (const float*)d_in[4];  const float* t_b1  = (const float*)d_in[5];
  const float* t_c2w = (const float*)d_in[6];  const float* t_c2b = (const float*)d_in[7];
  const float* t_g2  = (const float*)d_in[8];  const float* t_b2  = (const float*)d_in[9];
  const float* t_l1w = (const float*)d_in[10]; const float* t_l1b = (const float*)d_in[11];
  const float* t_l2w = (const float*)d_in[12]; const float* t_l2b = (const float*)d_in[13];
  const float* m_kw = (const float*)d_in[14];  const float* m_qw = (const float*)d_in[15];
  const float* m_vw = (const float*)d_in[16];
  const float* a_kw = (const float*)d_in[17];  const float* a_qw = (const float*)d_in[18];
  const float* a_vw = (const float*)d_in[19];
  const float* a_c1w = (const float*)d_in[20]; const float* a_c1b = (const float*)d_in[21];
  const float* a_g1  = (const float*)d_in[22]; const float* a_b1  = (const float*)d_in[23];
  const float* a_c2w = (const float*)d_in[24]; const float* a_c2b = (const float*)d_in[25];
  const float* a_g2  = (const float*)d_in[26]; const float* a_b2  = (const float*)d_in[27];
  const float* a_p1w = (const float*)d_in[28]; const float* a_p1b = (const float*)d_in[29];
  const float* a_p2w = (const float*)d_in[30]; const float* a_p2b = (const float*)d_in[31];

  float* out = (float*)d_out;          // [0,204800) pairs, [204800,205600) raw_off
  float* raw_off = out + 204800;

  float* ws = (float*)d_ws;
  size_t off = 0;
  auto alloc = [&](size_t n) { float* p = ws + off; off += (n + 63) & ~(size_t)63; return p; };
  float* h1    = alloc(15 * 64 * 392);    // locnet conv1 out
  float* lp1   = alloc(15 * 64 * 36);     // pooled1
  float* h2    = alloc(15 * 128 * 36);    // locnet conv2 out
  float* theta = alloc(32);
  float* s_al  = alloc(5 * 256 * 392);
  float* q_al  = alloc(10 * 256 * 392);
  float* si    = alloc(5 * 2048);
  float* keys1 = alloc(5 * 2048);
  float* qrs1  = alloc(5 * 2048);
  float* attn1 = alloc(5 * 64);
  float* vals1 = alloc(5 * 256 * 392);
  float* proto = alloc(5 * 256 * 392);
  float* sp    = alloc(5 * 2048);
  float* qp    = alloc(10 * 2048);
  float* keys2 = alloc(5 * 2048);
  float* qrs2  = alloc(10 * 2048);
  float* valsq = alloc(10 * 2048);
  float* valss = alloc(5 * 2048);
  float* attn2 = alloc(50 * 64);
  float* adjb  = alloc(50 * 2048);
  float* s_pr  = alloc(5 * 256 * 392);
  float* mh1   = alloc(50 * 128 * 392);
  float* mp1   = alloc(50 * 128 * 72);
  float* mh2   = alloc(50 * 128 * 72);
  float* mp2   = alloc(50 * 128 * 8);
  float* p1o   = alloc(50 * 64 * 8);
  (void)ws_size; (void)in_sizes; (void)n_in; (void)out_size;

  // ---- stage 1: TTM locnet (on 15 = 5 support + 10 query samples)
  conv3d_wmma<0><<<dim3(1, CDIV(15 * 392, 128)), 256, 0, stream>>>(
      t_c1w, t_c1b, t_g1, t_b1, support, query, nullptr, h1, 64, 256, 8, 7, 15);
  locnet_pool1_kernel<<<CDIV(15 * 64 * 36, 256), 256, 0, stream>>>(h1, lp1, 15 * 64 * 36);
  conv3d_wmma<1><<<dim3(2, CDIV(15 * 36, 128)), 256, 0, stream>>>(
      t_c2w, t_c2b, t_g2, t_b2, lp1, nullptr, nullptr, h2, 128, 64, 4, 3, 15);
  locnet_tail_kernel<<<15, 128, 0, stream>>>(h2, t_l1w, t_l1b, t_l2w, t_l2b, theta);
  warp_kernel<<<CDIV(15 * 256 * 392, 256), 256, 0, stream>>>(
      support, query, theta, s_al, q_al, 15 * 256 * 392);

  // ---- stage 2: insupport prototype (shot=1)
  mean_hw_kernel<<<CDIV(5 * 2048, 256), 256, 0, stream>>>(s_al, si, 5 * 2048, 49);
  mean_hw_kernel<<<CDIV(10 * 2048, 256), 256, 0, stream>>>(q_al, qp, 10 * 2048, 49);
  gemm_wmma<<<dim3(4, 1), 256, 0, stream>>>(m_kw, si, nullptr, keys1, 256, 256, 5, 8, 0);
  gemm_wmma<<<dim3(4, 1), 256, 0, stream>>>(m_qw, si, nullptr, qrs1, 256, 256, 5, 8, 0);
  gemm_wmma<<<dim3(4, CDIV(1960, 128)), 256, 0, stream>>>(m_vw, s_al, nullptr, vals1,
                                                          256, 256, 5, 392, 0);
  attn1_kernel<<<5, 64, 0, stream>>>(keys1, qrs1, attn1);
  proto_kernel<<<CDIV(5 * 256 * 392, 256), 256, 0, stream>>>(attn1, vals1, proto,
                                                             5 * 256 * 392);

  // ---- stage 3: ACM attention + adjustment (q_al_adj kept implicit)
  mean_hw_kernel<<<CDIV(5 * 2048, 256), 256, 0, stream>>>(proto, sp, 5 * 2048, 49);
  gemm_wmma<<<dim3(4, 1), 256, 0, stream>>>(a_kw, sp, nullptr, keys2, 256, 256, 5, 8, 0);
  gemm_wmma<<<dim3(4, 1), 256, 0, stream>>>(a_qw, qp, nullptr, qrs2, 256, 256, 10, 8, 0);
  gemm_wmma<<<dim3(4, 1), 256, 0, stream>>>(a_vw, qp, nullptr, valsq, 256, 256, 10, 8, 0);
  gemm_wmma<<<dim3(4, 1), 256, 0, stream>>>(a_vw, sp, nullptr, valss, 256, 256, 5, 8, 0);
  attn2_kernel<<<50, 64, 0, stream>>>(keys2, qrs2, attn2);
  adj_kernel<<<CDIV(50 * 2048, 256), 256, 0, stream>>>(attn2, valsq, adjb, 50 * 2048);
  spr_kernel<<<CDIV(5 * 256 * 392, 256), 256, 0, stream>>>(proto, valss, s_pr,
                                                           5 * 256 * 392);

  // ---- stage 4: mvnet (dominant GEMM: 128 x 13824 x 19600)
  conv3d_wmma<2><<<dim3(2, CDIV(50 * 392, 128)), 256, 0, stream>>>(
      a_c1w, a_c1b, a_g1, a_b1, s_pr, q_al, adjb, mh1, 128, 512, 8, 7, 50);
  mv_pool1_kernel<<<CDIV(50 * 128 * 72, 256), 256, 0, stream>>>(mh1, mp1, 50 * 128 * 72);
  conv3d_wmma<1><<<dim3(2, CDIV(50 * 72, 128)), 256, 0, stream>>>(
      a_c2w, a_c2b, a_g2, a_b2, mp1, nullptr, nullptr, mh2, 128, 128, 8, 3, 50);
  mv_pool2_kernel<<<CDIV(50 * 128 * 8, 256), 256, 0, stream>>>(mh2, mp2, 50 * 128 * 8);
  gemm_wmma<<<dim3(1, CDIV(400, 128)), 256, 0, stream>>>(a_p1w, mp2, a_p1b, p1o,
                                                         64, 128, 50, 8, 1);
  p2_kernel<<<CDIV(800, 256), 256, 0, stream>>>(a_p2w, a_p2b, p1o, raw_off, 800);

  // ---- stage 5: soft-mask pooling -> pairs output
  maskpool_kernel<<<400, 256, 0, stream>>>(s_pr, q_al, adjb, raw_off, out);
}
